// ProdAttention_76433237999705
// MI455X (gfx1250) — matrix operand order repaired and checked by reading, not compiled, not run on hardware
//
#include <hip/hip_runtime.h>

// ---------------------------------------------------------------------------
// Types / helpers
// ---------------------------------------------------------------------------
typedef unsigned short u16;
typedef unsigned int   u32;

typedef __attribute__((ext_vector_type(16))) __bf16 v16bf;
typedef __attribute__((ext_vector_type(8)))  float  v8f;

union Frag { v16bf v; uint4 q[2]; };

__device__ __forceinline__ u16 f2bf(float f) {
    union { float f; u32 u; } c; c.f = f;
    u32 u = c.u;
    u32 r = u + 0x7FFFu + ((u >> 16) & 1u);   // round-to-nearest-even
    return (u16)(r >> 16);
}

// order-preserving float <-> uint encoding for atomic min/max
__device__ __forceinline__ u32 fenc(float f) {
    union { float f; u32 u; } c; c.f = f;
    return (c.u & 0x80000000u) ? ~c.u : (c.u | 0x80000000u);
}
__device__ __forceinline__ float fdec(u32 e) {
    union { u32 u; float f; } c;
    c.u = (e & 0x80000000u) ? (e ^ 0x80000000u) : ~e;
    return c.f;
}

__device__ __forceinline__ v8f wmma_bf16(const Frag& a, const Frag& b, v8f c) {
    return __builtin_amdgcn_wmma_f32_16x16x32_bf16(
        /*neg_a=*/false, a.v, /*neg_b=*/false, b.v,
        /*c_mod=*/(short)0, c, /*reuse_a=*/false, /*reuse_b=*/false);
}

// A-fragment (16x32 bf16, M-major rows, K contiguous in memory):
//   lane: M = lane&15, half = lane>>4
//   elems 0..7  -> K = k0 + half*8 + e ; elems 8..15 -> K = k0+16+half*8+(e-8)
__device__ __forceinline__ Frag load_a(const u16* row, int k0, int half) {
    Frag f;
    f.q[0] = *(const uint4*)(row + k0 + half * 8);
    f.q[1] = *(const uint4*)(row + k0 + 16 + half * 8);
    return f;
}
// B-fragment (32x16 bf16): lane N = lane&15; elems 0..7 -> K = k0 + half*8 + e ; elems 8..15 -> K = k0+16+half*8+(e-8)
__device__ __forceinline__ Frag load_b(const u16* row, int k0, int half) {
    Frag f;
    f.q[0] = *(const uint4*)(row + k0 + half * 8);
    f.q[1] = *(const uint4*)(row + k0 + 16 + half * 8);
    return f;
}

// ---------------------------------------------------------------------------
// 0) fp32 -> bf16 convert (weights)
// ---------------------------------------------------------------------------
__global__ void cvt_bf16_kernel(const float* __restrict__ src, u16* __restrict__ dst, int n) {
    int i = blockIdx.x * blockDim.x + threadIdx.x;
    if (i < n) dst[i] = f2bf(src[i]);
}

// 0b) init global min/max encodings (re-run every launch: deterministic)
__global__ void init_minmax_kernel(u32* mm) {
    mm[0] = 0xFFFFFFFFu;  // running min (encoded)
    mm[1] = 0x00000000u;  // running max (encoded)
}

// ---------------------------------------------------------------------------
// 1) LayerNorm: x[4096,1024] -> xn bf16
// ---------------------------------------------------------------------------
__global__ void ln_kernel(const float* __restrict__ x,
                          const float* __restrict__ gamma,
                          const float* __restrict__ beta,
                          u16* __restrict__ xnh) {
    const int row = blockIdx.x;            // 0..4095
    const int t   = threadIdx.x;           // 0..255
    const float* xr = x + (size_t)row * 1024;
    float4 v = ((const float4*)xr)[t];

    __shared__ float s1[256], s2[256];
    s1[t] = v.x + v.y + v.z + v.w;
    s2[t] = v.x * v.x + v.y * v.y + v.z * v.z + v.w * v.w;
    __syncthreads();
    for (int st = 128; st > 0; st >>= 1) {
        if (t < st) { s1[t] += s1[t + st]; s2[t] += s2[t + st]; }
        __syncthreads();
    }
    const float mu  = s1[0] * (1.0f / 1024.0f);
    const float var = s2[0] * (1.0f / 1024.0f) - mu * mu;  // biased var (torch)
    const float inv = rsqrtf(var + 1e-5f);

    const float e[4] = {v.x, v.y, v.z, v.w};
    #pragma unroll
    for (int k = 0; k < 4; ++k) {
        int col = t * 4 + k;
        float y = (e[k] - mu) * inv * gamma[col] + beta[col];
        xnh[(size_t)row * 1024 + col] = f2bf(y);
    }
}

// ---------------------------------------------------------------------------
// route one GEMM result element into q (bf16 * scale) / k (bf16) / v (fp32)
// layout [b,h,n,d]; branch is uniform per 16-column sub-tile.
// ---------------------------------------------------------------------------
__device__ __forceinline__ void route_qkv(int grow, int col, float val,
                                          u16* __restrict__ qh,
                                          u16* __restrict__ kh,
                                          float* __restrict__ vf) {
    const int bb = grow >> 10, nn = grow & 1023;
    if (col < 1024) {                 // q (scaled by d^-0.5 = 1/8)
        int hh = col >> 6, dd = col & 63;
        qh[((size_t)(bb * 16 + hh) * 1024 + nn) * 64 + dd] = f2bf(val * 0.125f);
    } else if (col < 2048) {          // k
        int c = col - 1024, hh = c >> 6, dd = c & 63;
        kh[((size_t)(bb * 16 + hh) * 1024 + nn) * 64 + dd] = f2bf(val);
    } else {                          // v (fp32, exact for gather-sum)
        int c = col - 2048, hh = c >> 6, dd = c & 63;
        vf[((size_t)(bb * 16 + hh) * 1024 + nn) * 64 + dd] = val;
    }
}

// ---------------------------------------------------------------------------
// 2) QKV GEMM (bf16 WMMA): qkv[i,j] = sum_k xn[i,k] * Wqkv[j,k]
//    Each wave computes a 32x32 macro-tile (2 A frags x 2 B frags -> 4 WMMAs
//    per k-step => 1.0 fragment-load per WMMA, 2x the arithmetic intensity
//    of a 16x16-per-wave scheme).
//    grid (24, 128), block 128 (4 waves); wave w -> N32-tile blockIdx.x*4+w
// ---------------------------------------------------------------------------
__global__ void qkv_gemm_kernel(const u16* __restrict__ xnh,
                                const u16* __restrict__ wq,
                                u16* __restrict__ qh,
                                u16* __restrict__ kh,
                                float* __restrict__ vf) {
    const int wave = threadIdx.x >> 5;
    const int lane = threadIdx.x & 31;
    const int i0   = blockIdx.y * 32;
    const int j0   = (blockIdx.x * 4 + wave) * 32;
    const int m    = lane & 15;
    const int half = lane >> 4;

    const u16* arow0 = xnh + (size_t)(i0 + m) * 1024;
    const u16* arow1 = xnh + (size_t)(i0 + 16 + m) * 1024;
    const u16* brow0 = wq  + (size_t)(j0 + m) * 1024;
    const u16* brow1 = wq  + (size_t)(j0 + 16 + m) * 1024;

    v8f acc[2][2] = {{{}, {}}, {{}, {}}};
    for (int k0 = 0; k0 < 1024; k0 += 32) {
        if (k0 + 32 < 1024) {
            __builtin_prefetch(brow0 + k0 + 32, 0, 0);
            __builtin_prefetch(brow1 + k0 + 32, 0, 0);
        }
        Frag a0 = load_a(arow0, k0, half);
        Frag a1 = load_a(arow1, k0, half);
        Frag b0 = load_b(brow0, k0, half);
        Frag b1 = load_b(brow1, k0, half);
        acc[0][0] = wmma_bf16(a0, b0, acc[0][0]);
        acc[0][1] = wmma_bf16(a0, b1, acc[0][1]);
        acc[1][0] = wmma_bf16(a1, b0, acc[1][0]);
        acc[1][1] = wmma_bf16(a1, b1, acc[1][1]);
    }

    #pragma unroll
    for (int mt = 0; mt < 2; ++mt) {
        #pragma unroll
        for (int nt = 0; nt < 2; ++nt) {
            const int col = j0 + nt * 16 + (lane & 15);
            #pragma unroll
            for (int r = 0; r < 8; ++r) {
                const int grow = i0 + mt * 16 + half * 8 + r;   // b*1024+n
                route_qkv(grow, col, acc[mt][nt][r], qh, kh, vf);
            }
        }
    }
}

// ---------------------------------------------------------------------------
// 3) Attention: per (bh, 16-row tile): sim tile (WMMA) -> LDS, top-32 per row
//    (softmax skipped: monotonic => identical indices), gather-sum v,
//    atomic global min/max.  grid (64, 64), block 256 (8 waves).
// ---------------------------------------------------------------------------
__global__ void attn_kernel(const u16* __restrict__ qh,
                            const u16* __restrict__ kh,
                            const float* __restrict__ vf,
                            float* __restrict__ xsum,
                            u32* __restrict__ minmax) {
    __shared__ float sim[16 * 1024];      // 64 KB tile of logits
    const int bh   = blockIdx.y;          // b*16+h
    const int i0   = blockIdx.x * 16;
    const int wave = threadIdx.x >> 5;
    const int lane = threadIdx.x & 31;
    const int m    = lane & 15;
    const int half = lane >> 4;

    const u16* qbase = qh + ((size_t)bh * 1024 + i0) * 64;
    const u16* kbase = kh + (size_t)bh * 1024 * 64;

    // A fragments for the full K=64 (two k-steps), reused across all N-tiles
    const u16* arow = qbase + m * 64;
    Frag a0 = load_a(arow, 0, half);
    Frag a1 = load_a(arow, 32, half);

    for (int jt = wave; jt < 64; jt += 8) {
        const int j0 = jt * 16;
        const u16* brow = kbase + (size_t)(j0 + m) * 64;
        Frag b0 = load_b(brow, 0, half);
        Frag b1 = load_b(brow, 32, half);
        v8f acc = {};
        acc = wmma_bf16(a0, b0, acc);
        acc = wmma_bf16(a1, b1, acc);
        #pragma unroll
        for (int r = 0; r < 8; ++r)
            sim[(half * 8 + r) * 1024 + j0 + m] = acc[r];
    }
    __syncthreads();

    // top-32 selection + gather: each wave owns rows {wave, wave+8}
    for (int rr = 0; rr < 2; ++rr) {
        const int lrow = wave + rr * 8;
        float* srow = sim + lrow * 1024;
        float o0 = 0.0f, o1 = 0.0f;       // lane owns d = 2*lane, 2*lane+1
        for (int t = 0; t < 32; ++t) {
            float bm = -__builtin_inff(); int bi = 0x7FFFFFFF;
            for (int c = lane; c < 1024; c += 32) {
                float v = srow[c];
                if (v > bm) { bm = v; bi = c; }
            }
            // wave argmax, lowest index wins ties (matches jax top_k)
            #pragma unroll
            for (int off = 16; off > 0; off >>= 1) {
                float om = __shfl_xor(bm, off);
                int   oi = __shfl_xor(bi, off);
                if (om > bm || (om == bm && oi < bi)) { bm = om; bi = oi; }
            }
            if (lane == 0) srow[bi] = -__builtin_inff();  // remove selected
            const float* vr = vf + ((size_t)bh * 1024 + bi) * 64;
            o0 += vr[lane * 2];
            o1 += vr[lane * 2 + 1];
        }
        const size_t orow = ((size_t)bh * 1024 + i0 + lrow) * 64;
        xsum[orow + lane * 2]     = o0;
        xsum[orow + lane * 2 + 1] = o1;

        float mn = fminf(o0, o1), mx = fmaxf(o0, o1);
        #pragma unroll
        for (int off = 16; off > 0; off >>= 1) {
            mn = fminf(mn, __shfl_xor(mn, off));
            mx = fmaxf(mx, __shfl_xor(mx, off));
        }
        if (lane == 0) {
            atomicMin(&minmax[0], fenc(mn));
            atomicMax(&minmax[1], fenc(mx));
        }
    }
}

// ---------------------------------------------------------------------------
// 4) normalize + exp + head-merge transpose + bf16:
//    [b,h,n,d] -> norm[b*1024+n, h*64+d]
// ---------------------------------------------------------------------------
__global__ void norm_kernel(const float* __restrict__ xsum,
                            const u32* __restrict__ minmax,
                            u16* __restrict__ normh) {
    const float mn  = fdec(minmax[0]);
    const float mx  = fdec(minmax[1]);
    const float inv = 1.0f / (mx - mn);
    int s = blockIdx.x * blockDim.x + threadIdx.x;   // 0..4M-1
    if (s >= 4 * 16 * 1024 * 64) return;
    const int d = s & 63;
    const int n = (s >> 6) & 1023;
    const int h = (s >> 16) & 15;
    const int b = s >> 20;
    const float y = expf((xsum[s] - mn) * inv);
    normh[(((size_t)b * 1024 + n) << 10) + h * 64 + d] = f2bf(y);
}

// ---------------------------------------------------------------------------
// 5) Output GEMM: out[i,j] = sum_k norm[i,k]*Wout[j,k] + b_out[j]
//    32x32 macro-tile per wave; grid (8, 128), block 128 (4 waves)
// ---------------------------------------------------------------------------
__global__ void out_gemm_kernel(const u16* __restrict__ normh,
                                const u16* __restrict__ wout,
                                const float* __restrict__ bout,
                                float* __restrict__ out) {
    const int wave = threadIdx.x >> 5;
    const int lane = threadIdx.x & 31;
    const int i0   = blockIdx.y * 32;
    const int j0   = (blockIdx.x * 4 + wave) * 32;
    const int m    = lane & 15;
    const int half = lane >> 4;

    const u16* arow0 = normh + (size_t)(i0 + m) * 1024;
    const u16* arow1 = normh + (size_t)(i0 + 16 + m) * 1024;
    const u16* brow0 = wout  + (size_t)(j0 + m) * 1024;
    const u16* brow1 = wout  + (size_t)(j0 + 16 + m) * 1024;

    v8f acc[2][2] = {{{}, {}}, {{}, {}}};
    for (int k0 = 0; k0 < 1024; k0 += 32) {
        if (k0 + 32 < 1024) {
            __builtin_prefetch(brow0 + k0 + 32, 0, 0);
            __builtin_prefetch(brow1 + k0 + 32, 0, 0);
        }
        Frag a0 = load_a(arow0, k0, half);
        Frag a1 = load_a(arow1, k0, half);
        Frag b0 = load_b(brow0, k0, half);
        Frag b1 = load_b(brow1, k0, half);
        acc[0][0] = wmma_bf16(a0, b0, acc[0][0]);
        acc[0][1] = wmma_bf16(a0, b1, acc[0][1]);
        acc[1][0] = wmma_bf16(a1, b0, acc[1][0]);
        acc[1][1] = wmma_bf16(a1, b1, acc[1][1]);
    }

    #pragma unroll
    for (int mt = 0; mt < 2; ++mt) {
        #pragma unroll
        for (int nt = 0; nt < 2; ++nt) {
            const int col  = j0 + nt * 16 + (lane & 15);
            const float bias = bout[col];
            #pragma unroll
            for (int r = 0; r < 8; ++r) {
                const int row = i0 + mt * 16 + half * 8 + r;
                out[(size_t)row * 1024 + col] = acc[mt][nt][r] + bias;
            }
        }
    }
}

// ---------------------------------------------------------------------------
// Launch
// ---------------------------------------------------------------------------
extern "C" void kernel_launch(void* const* d_in, const int* in_sizes, int n_in,
                              void* d_out, int out_size, void* d_ws, size_t ws_size,
                              hipStream_t stream) {
    const float* x     = (const float*)d_in[0];
    const float* gamma = (const float*)d_in[1];
    const float* beta  = (const float*)d_in[2];
    const float* w_qkv = (const float*)d_in[3];
    const float* w_out = (const float*)d_in[4];
    const float* b_out = (const float*)d_in[5];
    // d_in[6] = topk (32) — fixed in-kernel

    // workspace layout (~72 MB)
    char* ws = (char*)d_ws;
    u16*   xnh    = (u16*)ws;                 ws += (size_t)4096 * 1024 * 2;    //  8 MB
    u16*   qh     = (u16*)ws;                 ws += (size_t)64 * 1024 * 64 * 2; //  8 MB
    u16*   kh     = (u16*)ws;                 ws += (size_t)64 * 1024 * 64 * 2; //  8 MB
    float* vf     = (float*)ws;               ws += (size_t)64 * 1024 * 64 * 4; // 16 MB
    u16*   wqkvh  = (u16*)ws;                 ws += (size_t)3072 * 1024 * 2;    //  6 MB
    u16*   wouth  = (u16*)ws;                 ws += (size_t)1024 * 1024 * 2;    //  2 MB
    float* xsum   = (float*)ws;               ws += (size_t)64 * 1024 * 64 * 4; // 16 MB
    u16*   normh  = (u16*)ws;                 ws += (size_t)4096 * 1024 * 2;    //  8 MB
    u32*   minmax = (u32*)ws;                                                  //  8 B

    // 0) weight conversions + min/max init
    cvt_bf16_kernel<<<(3072 * 1024 + 255) / 256, 256, 0, stream>>>(w_qkv, wqkvh, 3072 * 1024);
    cvt_bf16_kernel<<<(1024 * 1024 + 255) / 256, 256, 0, stream>>>(w_out, wouth, 1024 * 1024);
    init_minmax_kernel<<<1, 1, 0, stream>>>(minmax);

    // 1) LayerNorm
    ln_kernel<<<4096, 256, 0, stream>>>(x, gamma, beta, xnh);

    // 2) QKV projection (bf16 WMMA, 32x32 per wave)
    qkv_gemm_kernel<<<dim3(24, 128), 128, 0, stream>>>(xnh, wqkvh, qh, kh, vf);

    // 3) attention: sim tile + top-32 + gather-sum + global min/max
    attn_kernel<<<dim3(64, 64), 256, 0, stream>>>(qh, kh, vf, xsum, minmax);

    // 4) normalize/exp + merge heads
    norm_kernel<<<(4 * 16 * 1024 * 64 + 255) / 256, 256, 0, stream>>>(xsum, minmax, normh);

    // 5) output projection (bf16 WMMA, 32x32 per wave) + bias -> d_out
    out_gemm_kernel<<<dim3(8, 128), 128, 0, stream>>>(normh, wouth, b_out, (float*)d_out);
}
